// HierDDts_31817117728851
// MI455X (gfx1250) — matrix operand order repaired and checked by reading, not compiled, not run on hardware
//
#include <hip/hip_runtime.h>

#define B_    8
#define S_    1024
#define IN_   64
#define OUT_  128
#define NB_   6
#define LN_EPS 1e-5f

typedef __attribute__((ext_vector_type(16))) _Float16 v16h;
typedef __attribute__((ext_vector_type(8)))  _Float16 v8h;
typedef __attribute__((ext_vector_type(8)))  float    v8f;

union V16 { v16h v; struct { v8h lo, hi; } h; };

// ---------------------------------------------------------------------------
// Phase A: one block per row r = b*S + s  (8192 blocks x 128 threads)
//   x_t[i]    = dot(x[r,:], M[i,:])          (i = threadIdx.x)
//   residual  = dot(x[r,:], resW[i,:])  -> written to out (re-inits d_out)
//   LayerNorm over the 128 x_t values -> x_n stored as f16 in workspace
// ---------------------------------------------------------------------------
__global__ __launch_bounds__(OUT_) void hdd_phaseA(
    const float* __restrict__ x, const float* __restrict__ M,
    const float* __restrict__ resW, const float* __restrict__ gamma,
    const float* __restrict__ beta, _Float16* __restrict__ xn,
    float* __restrict__ out)
{
    __shared__ float xrow[IN_];
    __shared__ float red[OUT_];

    const int row = blockIdx.x;
    const int tid = threadIdx.x;

    if (tid < IN_) xrow[tid] = x[(size_t)row * IN_ + tid];
    __syncthreads();

    const float* mR = M    + (size_t)tid * IN_;
    const float* rR = resW + (size_t)tid * IN_;
    float acc = 0.f, accr = 0.f;
#pragma unroll
    for (int k = 0; k < IN_; ++k) {
        const float xv = xrow[k];
        acc  = fmaf(xv, mR[k], acc);
        accr = fmaf(xv, rR[k], accr);
    }

    // mean
    red[tid] = acc;
    __syncthreads();
    for (int off = OUT_ / 2; off > 0; off >>= 1) {
        if (tid < off) red[tid] += red[tid + off];
        __syncthreads();
    }
    const float mu = red[0] * (1.f / OUT_);
    __syncthreads();

    // variance
    const float d = acc - mu;
    red[tid] = d * d;
    __syncthreads();
    for (int off = OUT_ / 2; off > 0; off >>= 1) {
        if (tid < off) red[tid] += red[tid + off];
        __syncthreads();
    }
    const float var = red[0] * (1.f / OUT_);

    const float xnv = d * rsqrtf(var + LN_EPS) * gamma[tid] + beta[tid];

    xn [(size_t)row * OUT_ + tid] = (_Float16)xnv;
    out[(size_t)row * OUT_ + tid] = accr;        // plain store: re-initializes d_out
}

// ---------------------------------------------------------------------------
// Phase B: one block per position s (1024 blocks x 256 threads = 8 waves)
//   1) build W[s] (128x128, f16) in LDS:
//        W[i,j] = sum_g P[i,j,g] * cos(2*pi*s / (idx*6 + g + 2)),  idx = i*128+j
//      hardware V_COS takes revolutions: cos(2*pi*f) == v_cos(f)
//   2) A tile = x_n rows for batches 0..7 at this s, padded to 16 rows
//   3) per wave w: D(16x16) = A(16x128) x B(128x16), B[k][n] = W[16w+n][k]
//      via 4 chained v_wmma_f32_16x16x32_f16; accumulate into d_out
// ---------------------------------------------------------------------------
__global__ __launch_bounds__(256) void hdd_phaseB(
    const float* __restrict__ P, const _Float16* __restrict__ xn,
    float* __restrict__ out)
{
    __shared__ _Float16 Wls[OUT_ * OUT_];   // 32 KB
    __shared__ _Float16 Als[16 * OUT_];     //  4 KB

    const int s   = blockIdx.x;
    const int tid = threadIdx.x;
    const float sf = (float)s;

    // --- 1) position-dependent mixing matrix W[s] -> LDS (f16) -------------
    const int base = tid * 64;                       // 256 threads * 64 = 16384
    __builtin_prefetch(P + (size_t)base * NB_, 0, 0); // global_prefetch_b8
    for (int t = 0; t < 64; ++t) {
        const int idx = base + t;
        const float* pp = P + (size_t)idx * NB_;
        const float pbase = (float)(idx * NB_ + 2);
        float acc = 0.f;
#pragma unroll
        for (int g = 0; g < NB_; ++g) {
            const float rp = __builtin_amdgcn_rcpf(pbase + (float)g);
            const float ph = __builtin_amdgcn_fractf(sf * rp);   // in [0,1)
            acc = fmaf(pp[g], __builtin_amdgcn_cosf(ph), acc);   // v_cos_f32
        }
        Wls[idx] = (_Float16)acc;
    }

    // --- 2) A tile: rows 0..7 = x_n[b, s, :], rows 8..15 = 0 ---------------
    for (int idx = tid; idx < 16 * OUT_; idx += 256) {
        const int r = idx >> 7, c = idx & (OUT_ - 1);
        Als[idx] = (r < B_) ? xn[((size_t)r * S_ + s) * OUT_ + c] : (_Float16)0.f;
    }
    __syncthreads();

    // --- 3) WMMA: wave w owns output columns [16w, 16w+16) -----------------
    const int wave  = tid >> 5;
    const int lane  = tid & 31;
    const int lhalf = lane >> 4;      // 0: lanes 0-15, 1: lanes 16-31
    const int lrow  = lane & 15;

    v8f c = {};
#pragma unroll
    for (int kc = 0; kc < 4; ++kc) {
        // A fragment (16-bit A 16x32 layout): row = lrow;
        //   lanes 0-15 : K = kc*32 + {0..7, 16..23}
        //   lanes 16-31: K = kc*32 + {8..15, 24..31}
        const _Float16* ap = &Als[lrow * OUT_ + kc * 32 + lhalf * 8];
        V16 a;
        a.h.lo = *(const v8h*)(ap);
        a.h.hi = *(const v8h*)(ap + 16);

        // B fragment (16-bit B 32x16 layout): column n = lrow, B[k][n] = W[n][k]
        //   lanes 0-15 : K = kc*32 + {0..7, 16..23};  lanes 16-31: K = kc*32 + {8..15, 24..31}
        const _Float16* bp = &Wls[(wave * 16 + lrow) * OUT_ + kc * 32 + lhalf * 8];
        V16 b;
        b.h.lo = *(const v8h*)(bp);
        b.h.hi = *(const v8h*)(bp + 16);

        c = __builtin_amdgcn_wmma_f32_16x16x32_f16(
                /*neg_a=*/false, a.v, /*neg_b=*/false, b.v,
                /*c_mod=*/(short)0, c, /*reuse_a=*/false, /*reuse_b=*/false);
    }

    // --- 4) accumulate onto residual already in d_out ----------------------
    // C/D layout: lanes 0-15 VGPR r holds M=r (batches 0..7), col = lane
    if (lhalf == 0) {
        const int i = wave * 16 + lrow;
#pragma unroll
        for (int r = 0; r < B_; ++r) {
            const size_t o = ((size_t)r * S_ + s) * OUT_ + i;
            out[o] += c[r];
        }
    }
}

extern "C" void kernel_launch(void* const* d_in, const int* in_sizes, int n_in,
                              void* d_out, int out_size, void* d_ws, size_t ws_size,
                              hipStream_t stream) {
    const float* seq   = (const float*)d_in[0];
    const float* M     = (const float*)d_in[1];
    const float* P     = (const float*)d_in[2];
    const float* resW  = (const float*)d_in[3];
    const float* gamma = (const float*)d_in[4];
    const float* beta  = (const float*)d_in[5];
    float*    out = (float*)d_out;
    _Float16* xn  = (_Float16*)d_ws;            // 8192*128*2 = 2 MB scratch

    hdd_phaseA<<<B_ * S_, OUT_, 0, stream>>>(seq, M, resW, gamma, beta, xn, out);
    hdd_phaseB<<<S_, 256, 0, stream>>>(P, xn, out);
}
